// TemporalAudioEncoder_61349312856343
// MI455X (gfx1250) — matrix operand order repaired and checked by reading, not compiled, not run on hardware
//
#include <hip/hip_runtime.h>
#include <hip/hip_bf16.h>

typedef __attribute__((ext_vector_type(16))) _Float16 v16h;
typedef __attribute__((ext_vector_type(8)))  float    v8f;
typedef __attribute__((ext_vector_type(4)))  unsigned u32x4;
typedef __attribute__((ext_vector_type(8)))  int      i32x8;
typedef __attribute__((ext_vector_type(4)))  int      i32x4;

union V16H { v16h v; unsigned u[8]; _Float16 h[16]; };

#define EMB 512
#define HEADS 8
#define HEAD_DIM 64
#define FFN_DIM 2048
#define TLEN 2048
#define BATCH 4
#define ROWS (BATCH * TLEN)   // 8192

#define LDSOFF(p) ((unsigned)(unsigned long long)(p))

__device__ __forceinline__ float gelu_tanh(float x) {
    const float c = 0.7978845608028654f;   // sqrt(2/pi)
    float x3 = x * x * x;
    return 0.5f * x * (1.0f + tanhf(c * (x + 0.044715f * x3)));
}

// ---------------------------------------------------------------------------
// TDM: async copy of a 2D f16 tile (tile_w x tile_h) from global into LDS.
// Rows of tile_w=64 halfs (128B) are padded by 8 halfs (16B) in LDS ->
// LDS row stride 72 halfs, matching the fragment-read layout.
// pad_interval code 4 = pad after every 32 DWORDs; pad_amount code 3 = 4 DWORDs.
// Descriptor per CDNA5 ISA. Tracked by TENSORcnt.
// This toolchain declares the 6-arg builtin: (g0, g1, g2, g3, g4, cpol).
// ---------------------------------------------------------------------------
__device__ __forceinline__ void tdm_load_tile_f16(
    unsigned lds_byte_addr, const _Float16* gsrc,
    int tensor_w, int tensor_h, int row_stride_elems,
    int tile_w, int tile_h)
{
    unsigned long long ga = (unsigned long long)gsrc;
    u32x4 g0;
    g0[0] = 1u;                                        // count=1 (valid), user mode
    g0[1] = lds_byte_addr;                             // LDS destination (bytes)
    g0[2] = (unsigned)(ga & 0xFFFFFFFFu);              // global_addr[31:0]
    g0[3] = (unsigned)((ga >> 32) & 0x01FFFFFFu)       // global_addr[56:32]
          | 0x80000000u;                               // type = 2 ("image")
    i32x8 g1;
    g1[0] = (int)((1u << 16)                           // data_size = 2 bytes
                | (1u << 20)                           // pad_enable
                | (4u << 22)                           // pad_interval: 32 DWORDs
                | (3u << 25));                         // pad_amount: 4 DWORDs
    g1[1] = (int)(((unsigned)tensor_w & 0xFFFFu) << 16);               // dim0[15:0]
    g1[2] = (int)(((unsigned)tensor_w >> 16)
                | (((unsigned)tensor_h & 0xFFFFu) << 16));             // dim0 hi | dim1 lo
    g1[3] = (int)(((unsigned)tensor_h >> 16)
                | ((unsigned)tile_w << 16));                           // dim1 hi | tile_dim0
    g1[4] = (int)(unsigned)tile_h;                                     // tile_dim1 (tile_dim2=0)
    g1[5] = (int)(unsigned)row_stride_elems;                           // dim0_stride[31:0]
    g1[6] = 0;                                                         // dim0_stride hi, dim1_stride lo
    g1[7] = 0;
    i32x4 z4 = {0, 0, 0, 0};                           // 2D: groups 2/3 unused
    i32x8 z8 = {0, 0, 0, 0, 0, 0, 0, 0};
    __builtin_amdgcn_tensor_load_to_lds(g0, g1, z4, z4, z8, 0);
}

// ---------------------------------------------------------------------------
// f32 -> f16 conversion (vectorized x4).  n4 = element_count / 4.
// ---------------------------------------------------------------------------
__global__ __launch_bounds__(256) void f32_to_f16_kernel(
    const float* __restrict__ src, _Float16* __restrict__ dst, int n4)
{
    int id = blockIdx.x * 256 + threadIdx.x;
    if (id < n4) {
        float4 f = ((const float4*)src)[id];
        union { _Float16 h4[4]; uint2 u2; } pk;
        pk.h4[0] = (_Float16)f.x; pk.h4[1] = (_Float16)f.y;
        pk.h4[2] = (_Float16)f.z; pk.h4[3] = (_Float16)f.w;
        ((uint2*)dst)[id] = pk.u2;
    }
}

// ---------------------------------------------------------------------------
// GEMM: Y[M,N] = act(X[M,K] @ W[N,K]^T + bias[N]);  X, W are f16 in global.
// 256 threads (8 waves). Tile 128(M) x 64(N), K-step 64, double-buffered LDS
// filled by TDM tensor_load_to_lds (wave 0 issues; TENSORcnt in-order lets
// s_wait_tensorcnt(2) release the previous stage while the next is in flight).
// Inner loop contains only fragment ds_loads + 8 WMMA per wave per stage.
// ---------------------------------------------------------------------------
template <int ACT, int OUT16>   // ACT: 0 none, 1 gelu; OUT16: f16 or f32 output
__global__ __launch_bounds__(256) void gemm_wmma(
    const _Float16* __restrict__ X, const _Float16* __restrict__ W,
    const float* __restrict__ bias, void* __restrict__ Yv,
    int M, int N, int K)
{
    __shared__ __align__(16) _Float16 Xs[2][128][72];
    __shared__ __align__(16) _Float16 Ws[2][64][72];

    const int tid  = threadIdx.x;
    const int wv   = tid >> 5;
    const int lane = tid & 31;
    const int hf   = lane >> 4;
    const int lm   = lane & 15;
    const int wm   = wv & 3;        // 0..3  (M direction)
    const int wn   = wv >> 2;       // 0..1  (N direction)
    const int bm   = blockIdx.x * 128;
    const int bn   = blockIdx.y * 64;
    const int S    = K >> 6;        // number of 64-wide K stages

    v8f acc[2][2] = {};

    if (wv == 0) {   // prologue: stage 0 in flight
        tdm_load_tile_f16(LDSOFF(&Xs[0][0][0]), X + (size_t)bm * K,
                          K, M - bm, K, 64, 128);
        tdm_load_tile_f16(LDSOFF(&Ws[0][0][0]), W + (size_t)bn * K,
                          K, N - bn, K, 64, 64);
    }

    for (int s = 0; s < S; ++s) {
        if (wv == 0) {
            if (s + 1 < S) {
                int kb = (s + 1) << 6;
                int buf = (s + 1) & 1;
                tdm_load_tile_f16(LDSOFF(&Xs[buf][0][0]), X + (size_t)bm * K + kb,
                                  K - kb, M - bm, K, 64, 128);
                tdm_load_tile_f16(LDSOFF(&Ws[buf][0][0]), W + (size_t)bn * K + kb,
                                  K - kb, N - bn, K, 64, 64);
                __builtin_amdgcn_s_wait_tensorcnt(2);   // stage s landed
            } else {
                __builtin_amdgcn_s_wait_tensorcnt(0);
            }
        }
        __syncthreads();   // stage s visible to all waves

        const int buf = s & 1;
        V16H a[2][2], b[2][2];
        #pragma unroll
        for (int t = 0; t < 2; ++t) {
            int row = wm * 32 + t * 16 + lm;
            #pragma unroll
            for (int kc = 0; kc < 2; ++kc)
                #pragma unroll
                for (int v = 0; v < 8; ++v) {
                    int k0 = kc * 32 + ((v >> 2) << 4) + (hf << 3) + ((v & 3) << 1);
                    a[t][kc].u[v] = *(const unsigned*)&Xs[buf][row][k0];
                }
        }
        #pragma unroll
        for (int t = 0; t < 2; ++t) {
            int col = wn * 32 + t * 16 + lm;
            #pragma unroll
            for (int kc = 0; kc < 2; ++kc)
                #pragma unroll
                for (int v = 0; v < 8; ++v) {
                    int k0 = kc * 32 + ((v >> 2) << 4) + (hf << 3) + ((v & 3) << 1);
                    b[t][kc].u[v] = *(const unsigned*)&Ws[buf][col][k0];
                }
        }
        #pragma unroll
        for (int tm = 0; tm < 2; ++tm)
            #pragma unroll
            for (int tn = 0; tn < 2; ++tn)
                #pragma unroll
                for (int kc = 0; kc < 2; ++kc)
                    acc[tm][tn] = __builtin_amdgcn_wmma_f32_16x16x32_f16(
                        false, a[tm][kc].v, false, b[tn][kc].v, (short)0,
                        acc[tm][tn], false, false);
        __syncthreads();   // all waves done reading buf before TDM reuses it
    }

    // epilogue: D layout -> lane: row = r + 8*hf, col = lm
    #pragma unroll
    for (int tm = 0; tm < 2; ++tm) {
        #pragma unroll
        for (int tn = 0; tn < 2; ++tn) {
            int gc = bn + wn * 32 + tn * 16 + lm;
            float bval = bias[gc];
            #pragma unroll
            for (int r = 0; r < 8; ++r) {
                int gr = bm + wm * 32 + tm * 16 + hf * 8 + r;
                if (gr < M) {
                    float val = acc[tm][tn][r] + bval;
                    if (ACT == 1) val = gelu_tanh(val);
                    if (OUT16) ((_Float16*)Yv)[(size_t)gr * N + gc] = (_Float16)val;
                    else       ((float*)Yv)[(size_t)gr * N + gc] = val;
                }
            }
        }
    }
}

// ---------------------------------------------------------------------------
// RoPE in-place on f16 [ROWS, EMB]; multiplies result by `mult`
// (1/sqrt(hd) folded in for Q). One block per row, one thread per pair.
// ---------------------------------------------------------------------------
__global__ __launch_bounds__(256) void rope_kernel(_Float16* __restrict__ X, float mult)
{
    int row = blockIdx.x;
    int p = threadIdx.x;            // pair index 0..255
    int i = p & 31;                 // pair within head (dims 2i, 2i+1)
    int t = row & (TLEN - 1);
    float ang = (float)t * __expf(-(float)(2 * i) * 0.14391156462911484f);
    float s, c;
    __sincosf(ang, &s, &c);
    unsigned* base = (unsigned*)(X + (size_t)row * EMB + p * 2);
    union { unsigned u; _Float16 h[2]; } pk;
    pk.u = *base;
    float x0 = (float)pk.h[0], x1 = (float)pk.h[1];
    pk.h[0] = (_Float16)((x0 * c - x1 * s) * mult);
    pk.h[1] = (_Float16)((x1 * c + x0 * s) * mult);
    *base = pk.u;
}

// ---------------------------------------------------------------------------
// Flash attention: grid (T/64, HEADS, BATCH), block 128 (4 waves).
// f16 Q/K/V in global. Q and K tiles loaded by TDM; V staged transposed
// manually (TDM cannot transpose). Online softmax: row max via shfl
// butterfly; row sums on the WMMA pipe (P x ones). f16 output.
// 1/sqrt(hd) was folded into Q by rope_kernel.
// ---------------------------------------------------------------------------
__global__ __launch_bounds__(128) void flash_attn(
    const _Float16* __restrict__ Q, const _Float16* __restrict__ Km,
    const _Float16* __restrict__ Vm, _Float16* __restrict__ O)
{
    __shared__ __align__(16) _Float16 Qs[64][72];
    __shared__ __align__(16) _Float16 Ks[64][72];
    __shared__ __align__(16) _Float16 Vt[64][72];       // transposed: [hd col][key]
    __shared__ __align__(16) _Float16 Ps[4][16][72];    // per-wave P staging

    const int tid  = threadIdx.x;
    const int wv   = tid >> 5;      // wave 0..3
    const int lane = tid & 31;
    const int hf   = lane >> 4;
    const int lm   = lane & 15;
    const int b    = blockIdx.z;
    const int h    = blockIdx.y;
    const int q0   = blockIdx.x * 64;
    const size_t rowb = (size_t)b * TLEN;
    const int colb = h * HEAD_DIM;

    if (wv == 0) {
        tdm_load_tile_f16(LDSOFF(&Qs[0][0]), Q + (rowb + q0) * EMB + colb,
                          EMB - colb, 64, EMB, 64, 64);
        __builtin_amdgcn_s_wait_tensorcnt(0);
    }
    __syncthreads();

    // Q A-fragments (two K-chunks over hd=64)
    V16H qa[2];
    #pragma unroll
    for (int kk = 0; kk < 2; ++kk)
        #pragma unroll
        for (int v = 0; v < 8; ++v) {
            int k0 = kk * 32 + ((v >> 2) << 4) + (hf << 3) + ((v & 3) << 1);
            qa[kk].u[v] = *(const unsigned*)&Qs[wv * 16 + lm][k0];
        }

    // all-ones f16 B fragment for row sums on the matrix pipe
    V16H ones;
    #pragma unroll
    for (int v = 0; v < 8; ++v) ones.u[v] = 0x3C003C00u;

    float mrow[8], lrow[8];
    #pragma unroll
    for (int r = 0; r < 8; ++r) { mrow[r] = -1e30f; lrow[r] = 0.f; }
    v8f oacc[4] = {};

    for (int j = 0; j < TLEN; j += 64) {
        __syncthreads();   // previous iteration's LDS reads done
        if (wv == 0)
            tdm_load_tile_f16(LDSOFF(&Ks[0][0]), Km + (rowb + j) * EMB + colb,
                              EMB - colb, 64, EMB, 64, 64);
        // stage V transposed: 1024 x uint2 (4 halfs) over 128 threads
        #pragma unroll
        for (int i = 0; i < 8; ++i) {
            int idx = tid + i * 128;
            int r = idx >> 4;
            int c = (idx & 15) << 2;
            union { uint2 u; _Float16 h4[4]; } pk;
            pk.u = *(const uint2*)(Vm + (rowb + j + r) * EMB + colb + c);
            Vt[c + 0][r] = pk.h4[0];
            Vt[c + 1][r] = pk.h4[1];
            Vt[c + 2][r] = pk.h4[2];
            Vt[c + 3][r] = pk.h4[3];
        }
        if (wv == 0) __builtin_amdgcn_s_wait_tensorcnt(0);
        __syncthreads();

        // S = Q K^T : 16 q rows x 64 keys (4 n-tiles x 2 hd-chunks)
        v8f s[4] = {};
        #pragma unroll
        for (int tn = 0; tn < 4; ++tn) {
            #pragma unroll
            for (int kk = 0; kk < 2; ++kk) {
                V16H bk;
                #pragma unroll
                for (int v = 0; v < 8; ++v) {
                    int k0 = kk * 32 + ((v >> 2) << 4) + (hf << 3) + ((v & 3) << 1);   // hd index
                    bk.u[v] = *(const unsigned*)&Ks[tn * 16 + lm][k0];
                }
                s[tn] = __builtin_amdgcn_wmma_f32_16x16x32_f16(
                    false, qa[kk].v, false, bk.v, (short)0, s[tn], false, false);
            }
        }

        // online softmax: max via shfl butterfly, exp elementwise
        float alpha[8];
        #pragma unroll
        for (int r = 0; r < 8; ++r) {
            float mx = fmaxf(fmaxf(s[0][r], s[1][r]), fmaxf(s[2][r], s[3][r]));
            #pragma unroll
            for (int off = 1; off < 16; off <<= 1)
                mx = fmaxf(mx, __shfl_xor(mx, off, 32));
            float mn = fmaxf(mrow[r], mx);
            alpha[r] = __expf(mrow[r] - mn);
            mrow[r] = mn;
            #pragma unroll
            for (int tn = 0; tn < 4; ++tn)
                s[tn][r] = __expf(s[tn][r] - mn);
        }

        // D-layout -> A-layout for P via per-wave LDS
        #pragma unroll
        for (int r = 0; r < 8; ++r) {
            int rr = hf * 8 + r;
            #pragma unroll
            for (int tn = 0; tn < 4; ++tn)
                Ps[wv][rr][tn * 16 + lm] = (_Float16)s[tn][r];
        }
        __syncthreads();

        V16H pa[2];
        #pragma unroll
        for (int kc = 0; kc < 2; ++kc)
            #pragma unroll
            for (int v = 0; v < 8; ++v) {
                int k0 = kc * 32 + ((v >> 2) << 4) + (hf << 3) + ((v & 3) << 1);
                pa[kc].u[v] = *(const unsigned*)&Ps[wv][lm][k0];
            }

        // row sums on matrix pipe: sacc[r] = sum_k P[row][k]
        v8f sacc = {};
        sacc = __builtin_amdgcn_wmma_f32_16x16x32_f16(
            false, pa[0].v, false, ones.v, (short)0, sacc, false, false);
        sacc = __builtin_amdgcn_wmma_f32_16x16x32_f16(
            false, pa[1].v, false, ones.v, (short)0, sacc, false, false);
        #pragma unroll
        for (int r = 0; r < 8; ++r)
            lrow[r] = lrow[r] * alpha[r] + sacc[r];

        // O = alpha*O + P V   (K-dim = 64 keys, 4 hd column tiles)
        #pragma unroll
        for (int t = 0; t < 4; ++t) {
            #pragma unroll
            for (int r = 0; r < 8; ++r) oacc[t][r] *= alpha[r];
            #pragma unroll
            for (int kc = 0; kc < 2; ++kc) {
                V16H bvf;
                #pragma unroll
                for (int v = 0; v < 8; ++v) {
                    int k0 = kc * 32 + ((v >> 2) << 4) + (hf << 3) + ((v & 3) << 1);   // key index
                    bvf.u[v] = *(const unsigned*)&Vt[t * 16 + lm][k0];
                }
                oacc[t] = __builtin_amdgcn_wmma_f32_16x16x32_f16(
                    false, pa[kc].v, false, bvf.v, (short)0, oacc[t], false, false);
            }
        }
    }

    #pragma unroll
    for (int t = 0; t < 4; ++t)
        #pragma unroll
        for (int r = 0; r < 8; ++r) {
            int rr = hf * 8 + r;
            float val = oacc[t][r] / lrow[r];
            O[(rowb + q0 + wv * 16 + rr) * EMB + colb + t * 16 + lm] = (_Float16)val;
        }
}

// ---------------------------------------------------------------------------
// LayerNorm(X (+ R)) * g + b -> optional f32 Y and/or f16 Yh.
// One wave per row (EMB=512).
// ---------------------------------------------------------------------------
__global__ __launch_bounds__(256) void add_ln_kernel(
    const float* __restrict__ X, const float* __restrict__ R,
    const float* __restrict__ g, const float* __restrict__ bb,
    float* __restrict__ Y, _Float16* __restrict__ Yh, int nrows)
{
    int wv = threadIdx.x >> 5, lane = threadIdx.x & 31;
    int row = blockIdx.x * 8 + wv;
    if (row >= nrows) return;
    const float* xr = X + (size_t)row * EMB;

    float v[16];
    float sum = 0.f;
    #pragma unroll
    for (int i = 0; i < 4; ++i) {
        int col = lane * 4 + i * 128;
        float4 f = *(const float4*)(xr + col);
        if (R) {
            float4 rr = *(const float4*)(R + (size_t)row * EMB + col);
            f.x += rr.x; f.y += rr.y; f.z += rr.z; f.w += rr.w;
        }
        v[4 * i] = f.x; v[4 * i + 1] = f.y; v[4 * i + 2] = f.z; v[4 * i + 3] = f.w;
        sum += f.x + f.y + f.z + f.w;
    }
    #pragma unroll
    for (int off = 1; off < 32; off <<= 1) sum += __shfl_xor(sum, off, 32);
    float mu = sum * (1.f / EMB);

    float vs = 0.f;
    #pragma unroll
    for (int i = 0; i < 16; ++i) { float d = v[i] - mu; vs += d * d; }
    #pragma unroll
    for (int off = 1; off < 32; off <<= 1) vs += __shfl_xor(vs, off, 32);
    float rstd = rsqrtf(vs * (1.f / EMB) + 1e-5f);

    #pragma unroll
    for (int i = 0; i < 4; ++i) {
        int col = lane * 4 + i * 128;
        float4 gg = *(const float4*)(g + col);
        float4 bv = *(const float4*)(bb + col);
        float4 o;
        o.x = (v[4 * i]     - mu) * rstd * gg.x + bv.x;
        o.y = (v[4 * i + 1] - mu) * rstd * gg.y + bv.y;
        o.z = (v[4 * i + 2] - mu) * rstd * gg.z + bv.z;
        o.w = (v[4 * i + 3] - mu) * rstd * gg.w + bv.w;
        if (Y) *(float4*)(Y + (size_t)row * EMB + col) = o;
        if (Yh) {
            union { _Float16 h4[4]; uint2 u2; } pk;
            pk.h4[0] = (_Float16)o.x; pk.h4[1] = (_Float16)o.y;
            pk.h4[2] = (_Float16)o.z; pk.h4[3] = (_Float16)o.w;
            *(uint2*)(Yh + (size_t)row * EMB + col) = pk.u2;
        }
    }
}

// ---------------------------------------------------------------------------
// Mean pool over T (f16 in, f32 accumulate, f16 out): out[b,d] = mean_t X[b,t,d]
// ---------------------------------------------------------------------------
__global__ __launch_bounds__(256) void pool_kernel(
    const _Float16* __restrict__ X, _Float16* __restrict__ out)
{
    int id = blockIdx.x * 256 + threadIdx.x;   // 0..2047
    int b = id >> 9, d = id & 511;
    const _Float16* p = X + ((size_t)b * TLEN) * EMB + d;
    float s = 0.f;
    for (int t = 0; t < TLEN; ++t) s += (float)p[(size_t)t * EMB];
    out[id] = (_Float16)(s * (1.0f / TLEN));
}

// ---------------------------------------------------------------------------
// Host launcher. Pytree (sorted-key) input order:
//  0: audio_embeddings ; 1,2: final_ln (g,b)
//  layer L base = 3 + 16L:
//    +0 ffn1.w +1 ffn1.b +2 ffn2.w +3 ffn2.b +4 k.w +5 k.b
//    +6 ln1.g +7 ln1.b +8 ln2.g +9 ln2.b +10 o.w +11 o.b
//    +12 q.w +13 q.b +14 v.w +15 v.b
//  35,36: mlp1 (w,b) ; 37,38: mlp2 (w,b)
// ---------------------------------------------------------------------------
extern "C" void kernel_launch(void* const* d_in, const int* in_sizes, int n_in,
                              void* d_out, int out_size, void* d_ws, size_t ws_size,
                              hipStream_t stream)
{
    (void)in_sizes; (void)n_in; (void)out_size; (void)ws_size;

    const float* x_in   = (const float*)d_in[0];
    const float* fin_g  = (const float*)d_in[1];
    const float* fin_b  = (const float*)d_in[2];
    const float* mlp1_w = (const float*)d_in[35];
    const float* mlp1_b = (const float*)d_in[36];
    const float* mlp2_w = (const float*)d_in[37];
    const float* mlp2_b = (const float*)d_in[38];

    const size_t SZ  = (size_t)ROWS * EMB;   // 4194304 elements
    const size_t WSZ = (size_t)EMB * EMB;    // 262144
    const size_t W1SZ = (size_t)FFN_DIM * EMB; // 1048576

    float* xws  = (float*)d_ws;              // f32 activations
    float* ybuf = xws + SZ;                  // f32 gemm outputs
    _Float16* xh   = (_Float16*)(ybuf + SZ); // f16 GEMM input copy
    _Float16* r16  = xh + SZ;                // 4*SZ reusable f16 region
    _Float16* qh   = r16;
    _Float16* kh   = r16 + SZ;
    _Float16* vh   = r16 + 2 * SZ;
    _Float16* ah   = r16 + 3 * SZ;
    _Float16* hid  = r16;                    // FFN hidden (reuses qkv/attn)
    _Float16* fin16  = qh;                   // post-final-LN (reuse after layers)
    _Float16* pooled = kh;                   // head buffers (reuse)
    _Float16* htmp   = kh + 4096;
    _Float16* wts  = r16 + 4 * SZ;           // f16 weights

    // f16 weight pointers
    _Float16 *wq16[2], *wk16[2], *wv16[2], *wo16[2], *w116[2], *w216[2];
    _Float16* wp = wts;
    for (int L = 0; L < 2; ++L) {
        wq16[L] = wp; wp += WSZ;
        wk16[L] = wp; wp += WSZ;
        wv16[L] = wp; wp += WSZ;
        wo16[L] = wp; wp += WSZ;
        w116[L] = wp; wp += W1SZ;
        w216[L] = wp; wp += W1SZ;
    }
    _Float16* mlp1_16 = wp; wp += WSZ;
    _Float16* mlp2_16 = wp;

    // one-time conversions (deterministic; run every launch)
    auto cvt = [&](const float* s, _Float16* dst, size_t n) {
        int n4 = (int)(n >> 2);
        f32_to_f16_kernel<<<(n4 + 255) / 256, 256, 0, stream>>>(s, dst, n4);
    };
    cvt(x_in, xh, SZ);
    for (int L = 0; L < 2; ++L) {
        int base = 3 + L * 16;
        cvt((const float*)d_in[base + 12], wq16[L], WSZ);
        cvt((const float*)d_in[base + 4],  wk16[L], WSZ);
        cvt((const float*)d_in[base + 14], wv16[L], WSZ);
        cvt((const float*)d_in[base + 10], wo16[L], WSZ);
        cvt((const float*)d_in[base + 0],  w116[L], W1SZ);
        cvt((const float*)d_in[base + 2],  w216[L], W1SZ);
    }
    cvt(mlp1_w, mlp1_16, WSZ);
    cvt(mlp2_w, mlp2_16, WSZ);

    const dim3 blk(256);
    const dim3 g1(ROWS / 128, EMB / 64);
    const dim3 g2(ROWS / 128, FFN_DIM / 64);
    const float* xcur = x_in;   // f32 residual stream

    for (int L = 0; L < 2; ++L) {
        int base = 3 + L * 16;
        const float* b1   = (const float*)d_in[base + 1];
        const float* b2   = (const float*)d_in[base + 3];
        const float* bk_  = (const float*)d_in[base + 5];
        const float* ln1g = (const float*)d_in[base + 6];
        const float* ln1b = (const float*)d_in[base + 7];
        const float* ln2g = (const float*)d_in[base + 8];
        const float* ln2b = (const float*)d_in[base + 9];
        const float* bo   = (const float*)d_in[base + 11];
        const float* bq   = (const float*)d_in[base + 13];
        const float* bvv  = (const float*)d_in[base + 15];

        gemm_wmma<0, 1><<<g1, blk, 0, stream>>>(xh, wq16[L], bq, qh, ROWS, EMB, EMB);
        gemm_wmma<0, 1><<<g1, blk, 0, stream>>>(xh, wk16[L], bk_, kh, ROWS, EMB, EMB);
        gemm_wmma<0, 1><<<g1, blk, 0, stream>>>(xh, wv16[L], bvv, vh, ROWS, EMB, EMB);
        rope_kernel<<<ROWS, 256, 0, stream>>>(qh, 0.125f);   // fold 1/sqrt(hd)
        rope_kernel<<<ROWS, 256, 0, stream>>>(kh, 1.0f);
        flash_attn<<<dim3(TLEN / 64, HEADS, BATCH), 128, 0, stream>>>(qh, kh, vh, ah);
        gemm_wmma<0, 0><<<g1, blk, 0, stream>>>(ah, wo16[L], bo, ybuf, ROWS, EMB, EMB);
        add_ln_kernel<<<ROWS / 8, 256, 0, stream>>>(ybuf, xcur, ln1g, ln1b, xws, xh, ROWS);
        gemm_wmma<1, 1><<<g2, blk, 0, stream>>>(xh, w116[L], b1, hid, ROWS, FFN_DIM, EMB);
        gemm_wmma<0, 0><<<g1, blk, 0, stream>>>(hid, w216[L], b2, ybuf, ROWS, EMB, FFN_DIM);
        add_ln_kernel<<<ROWS / 8, 256, 0, stream>>>(ybuf, xws, ln2g, ln2b, xws, xh, ROWS);
        xcur = xws;
    }

    add_ln_kernel<<<ROWS / 8, 256, 0, stream>>>(xws, nullptr, fin_g, fin_b,
                                                nullptr, fin16, ROWS);
    pool_kernel<<<8, 256, 0, stream>>>(fin16, pooled);
    gemm_wmma<1, 1><<<dim3(1, EMB / 64), blk, 0, stream>>>(pooled, mlp1_16, mlp1_b,
                                                           htmp, BATCH, EMB, EMB);
    gemm_wmma<0, 0><<<dim3(1, EMB / 64), blk, 0, stream>>>(htmp, mlp2_16, mlp2_b,
                                                           d_out, BATCH, EMB, EMB);
}
